// HeteroGAT_81071802679584
// MI455X (gfx1250) — matrix operand order repaired and checked by reading, not compiled, not run on hardware
//
#include <hip/hip_runtime.h>
#include <hip/hip_bf16.h>

#define H_ 4
#define C_ 32
#define NT_ 50000
#define NDAT_ 20000
#define NDEV_ 4
#define EDT_ 800000
#define EVT_ 200000
#define ETT_ 400000

typedef __attribute__((ext_vector_type(16))) _Float16 v16h;
typedef __attribute__((ext_vector_type(8)))  _Float16 h8;
typedef __attribute__((ext_vector_type(8)))  float    v8f;

// ---- order-preserving float<->uint encoding for atomic segment-max ----
// enc is monotonic; all encodings of real floats are > 0, so memset(0) == -inf init.
__device__ __forceinline__ unsigned fenc(float f) {
  unsigned u = __float_as_uint(f);
  return (u >> 31) ? ~u : (u | 0x80000000u);
}
__device__ __forceinline__ float fdec(unsigned u) {
  return __uint_as_float((u >> 31) ? (u & 0x7FFFFFFFu) : ~u);
}

// ---- fold attention vector into edge/dst weight:  F[k][h] = sum_c W[k, h*32+c] * A[h, c] ----
__global__ void k_fold(const float* __restrict__ W, const float* __restrict__ A,
                       float* __restrict__ F, int K) {
  int i = blockIdx.x * 256 + threadIdx.x;
  if (i >= K * 4) return;
  int k = i >> 2, h = i & 3;
  float acc = 0.f;
  #pragma unroll
  for (int c = 0; c < 32; ++c) acc += W[(size_t)k * 128 + h * 32 + c] * A[h * 32 + c];
  F[i] = acc;
}

// ---- small dense GEMM: Y[M,128] = X[M,K] @ W[K,128], K<=16, W staged in LDS ----
__global__ __launch_bounds__(256) void k_lingemm(const float* __restrict__ X,
                                                 const float* __restrict__ W,
                                                 float* __restrict__ Y, int M, int K) {
  __shared__ float w[16 * 128];
  for (int i = threadIdx.x; i < K * 128; i += 256) w[i] = W[i];
  __syncthreads();
  int col = threadIdx.x & 127;
  int m = blockIdx.x * 2 + (threadIdx.x >> 7);
  if (m >= M) return;
  float acc = 0.f;
  for (int k = 0; k < K; ++k) acc += X[(size_t)m * K + k] * w[k * 128 + col];
  Y[(size_t)m * 128 + col] = acc;
}

// ---- S[M,4] = X[M,K] @ F[K,4]  (folded dst-side attention) ----
__global__ void k_small(const float* __restrict__ X, const float* __restrict__ F,
                        float* __restrict__ S, int M, int K) {
  int m = blockIdx.x * 256 + threadIdx.x;
  if (m >= M) return;
  float a[4] = {0.f, 0.f, 0.f, 0.f};
  for (int k = 0; k < K; ++k) {
    float x = X[(size_t)m * K + k];
    #pragma unroll
    for (int h = 0; h < 4; ++h) a[h] += x * F[k * 4 + h];
  }
  #pragma unroll
  for (int h = 0; h < 4; ++h) S[(size_t)m * 4 + h] = a[h];
}

// ---- S[n,h] = sum_c Hs[n, h*32+c] * A[h,c]  (src-side attention from projected feats) ----
__global__ void k_hdot(const float* __restrict__ Hs, const float* __restrict__ A,
                       float* __restrict__ S, int N) {
  int i = blockIdx.x * 256 + threadIdx.x;
  if (i >= N * 4) return;
  int n = i >> 2, h = i & 3;
  const float* hp = Hs + (size_t)n * 128 + h * 32;
  const float* ap = A + h * 32;
  float acc = 0.f;
  #pragma unroll
  for (int c = 0; c < 32; ++c) acc += hp[c] * ap[c];
  S[i] = acc;
}

// ---- pass 1: per-edge attention logit -> atomic segment max over dst ----
__global__ void k_edge_max(const int* __restrict__ src, const int* __restrict__ dst,
                           const float* __restrict__ ea, int edim,
                           const float* __restrict__ wef,
                           const float* __restrict__ ssrc, const float* __restrict__ sdst,
                           unsigned* __restrict__ amax, int E) {
  int e = blockIdx.x * 256 + threadIdx.x;
  if (e >= E) return;
  int s = src[e], d = dst[e];
  float lg[4];
  #pragma unroll
  for (int h = 0; h < 4; ++h) lg[h] = ssrc[(size_t)s * 4 + h] + sdst[(size_t)d * 4 + h];
  const float* ap = ea + (size_t)e * edim;
  for (int k = 0; k < edim; ++k) {
    float a = ap[k];
    #pragma unroll
    for (int h = 0; h < 4; ++h) lg[h] += a * wef[k * 4 + h];
  }
  #pragma unroll
  for (int h = 0; h < 4; ++h) {
    float x = lg[h];
    x = x > 0.f ? x : 0.2f * x;
    atomicMax(&amax[(size_t)d * 4 + h], fenc(x));
  }
}

// ---- pass 2: recompute logit, exp, atomic den + unnormalized weighted scatter (wave/edge) ----
__global__ __launch_bounds__(256) void k_edge_acc(const int* __restrict__ src,
                                                  const int* __restrict__ dst,
                                                  const float* __restrict__ ea, int edim,
                                                  const float* __restrict__ wef,
                                                  const float* __restrict__ ssrc,
                                                  const float* __restrict__ sdst,
                                                  const unsigned* __restrict__ amax,
                                                  const float* __restrict__ hsrc,
                                                  float* __restrict__ outacc,
                                                  float* __restrict__ den, int E) {
  int wv = (blockIdx.x * 256 + threadIdx.x) >> 5;
  int lane = threadIdx.x & 31;
  if (wv >= E) return;
  int s = src[wv], d = dst[wv];
  float lg[4];
  #pragma unroll
  for (int h = 0; h < 4; ++h) lg[h] = ssrc[(size_t)s * 4 + h] + sdst[(size_t)d * 4 + h];
  const float* ap = ea + (size_t)wv * edim;
  for (int k = 0; k < edim; ++k) {
    float a = ap[k];
    #pragma unroll
    for (int h = 0; h < 4; ++h) lg[h] += a * wef[k * 4 + h];
  }
  float ex[4];
  #pragma unroll
  for (int h = 0; h < 4; ++h) {
    float x = lg[h];
    x = x > 0.f ? x : 0.2f * x;
    ex[h] = __expf(x - fdec(amax[(size_t)d * 4 + h]));
  }
  if (lane < 4) atomicAdd(&den[(size_t)d * 4 + lane], ex[lane]);
  const float* hp = hsrc + (size_t)s * 128;
  float* op = outacc + (size_t)d * 128;
  #pragma unroll
  for (int h = 0; h < 4; ++h) atomicAdd(&op[h * 32 + lane], hp[h * 32 + lane] * ex[h]);
}

// ---- conv3 variants: combined edge list = ETT real edges + NT self-loops (ea = per-node mean) ----
__global__ void k_edge3_max(const int* __restrict__ src, const int* __restrict__ dst,
                            const float* __restrict__ ea, const float* __restrict__ eamean,
                            const float* __restrict__ wef,
                            const float* __restrict__ ssrc, const float* __restrict__ sdst,
                            unsigned* __restrict__ amax) {
  int e = blockIdx.x * 256 + threadIdx.x;
  if (e >= ETT_ + NT_) return;
  int s, d; const float* ap;
  if (e < ETT_) { s = src[e]; d = dst[e]; ap = ea + (size_t)e * 4; }
  else          { s = d = e - ETT_;       ap = eamean + (size_t)s * 4; }
  float lg[4];
  #pragma unroll
  for (int h = 0; h < 4; ++h) lg[h] = ssrc[(size_t)s * 4 + h] + sdst[(size_t)d * 4 + h];
  #pragma unroll
  for (int k = 0; k < 4; ++k) {
    float a = ap[k];
    #pragma unroll
    for (int h = 0; h < 4; ++h) lg[h] += a * wef[k * 4 + h];
  }
  #pragma unroll
  for (int h = 0; h < 4; ++h) {
    float x = lg[h];
    x = x > 0.f ? x : 0.2f * x;
    atomicMax(&amax[(size_t)d * 4 + h], fenc(x));
  }
}

__global__ __launch_bounds__(256) void k_edge3_acc(const int* __restrict__ src,
                                                   const int* __restrict__ dst,
                                                   const float* __restrict__ ea,
                                                   const float* __restrict__ eamean,
                                                   const float* __restrict__ wef,
                                                   const float* __restrict__ ssrc,
                                                   const float* __restrict__ sdst,
                                                   const unsigned* __restrict__ amax,
                                                   const float* __restrict__ hs3,
                                                   float* __restrict__ outacc,
                                                   float* __restrict__ den) {
  int wv = (blockIdx.x * 256 + threadIdx.x) >> 5;
  int lane = threadIdx.x & 31;
  if (wv >= ETT_ + NT_) return;
  int s, d; const float* ap;
  if (wv < ETT_) { s = src[wv]; d = dst[wv]; ap = ea + (size_t)wv * 4; }
  else           { s = d = wv - ETT_;        ap = eamean + (size_t)s * 4; }
  float lg[4];
  #pragma unroll
  for (int h = 0; h < 4; ++h) lg[h] = ssrc[(size_t)s * 4 + h] + sdst[(size_t)d * 4 + h];
  #pragma unroll
  for (int k = 0; k < 4; ++k) {
    float a = ap[k];
    #pragma unroll
    for (int h = 0; h < 4; ++h) lg[h] += a * wef[k * 4 + h];
  }
  float ex[4];
  #pragma unroll
  for (int h = 0; h < 4; ++h) {
    float x = lg[h];
    x = x > 0.f ? x : 0.2f * x;
    ex[h] = __expf(x - fdec(amax[(size_t)d * 4 + h]));
  }
  if (lane < 4) atomicAdd(&den[(size_t)d * 4 + lane], ex[lane]);
  const float* hp = hs3 + (size_t)s * 128;
  float* op = outacc + (size_t)d * 128;
  #pragma unroll
  for (int h = 0; h < 4; ++h) atomicAdd(&op[h * 32 + lane], hp[h * 32 + lane] * ex[h]);
}

// ---- normalize + bias + relu, store f16 into tasks[:, coloff:coloff+128] ----
__global__ void k_norm(const float* __restrict__ acc, const float* __restrict__ den,
                       const float* __restrict__ b, _Float16* __restrict__ t16, int coloff) {
  int i = blockIdx.x * 256 + threadIdx.x;
  if (i >= NT_ * 128) return;
  int n = i >> 7, j = i & 127, h = j >> 5;
  float v = acc[i] / (den[(size_t)n * 4 + h] + 1e-16f) + b[j];
  v = v > 0.f ? v : 0.f;
  t16[(size_t)n * 256 + coloff + j] = (_Float16)v;
}

// ---- self-loop edge-attr mean: accumulate ----
__global__ void k_easum(const int* __restrict__ dst, const float* __restrict__ ea,
                        float* __restrict__ easum, float* __restrict__ cnt, int E) {
  int e = blockIdx.x * 256 + threadIdx.x;
  if (e >= E) return;
  int d = dst[e];
  atomicAdd(&cnt[d], 1.f);
  #pragma unroll
  for (int j = 0; j < 4; ++j) atomicAdd(&easum[(size_t)d * 4 + j], ea[(size_t)e * 4 + j]);
}
__global__ void k_eamean(const float* __restrict__ easum, const float* __restrict__ cnt,
                         float* __restrict__ eamean) {
  int n = blockIdx.x * 256 + threadIdx.x;
  if (n >= NT_) return;
  float inv = 1.f / fmaxf(cnt[n], 1.f);
  #pragma unroll
  for (int j = 0; j < 4; ++j) eamean[(size_t)n * 4 + j] = easum[(size_t)n * 4 + j] * inv;
}

// ---- pre-swizzle W3 (f32 [256,128]) into WMMA B-operand lane layout (f16) ----
// B 16-bit 32x16: lane = N column (lane&15); lane group g=(lane>>4): elements 0..7 -> K 8g..8g+7, 8..15 -> K 16+8g..16+8g+7.
__global__ void k_pack_w3(const float* __restrict__ W3, _Float16* __restrict__ bswz) {
  int idx = blockIdx.x * 256 + threadIdx.x;       // 8 kk * 8 ntile * 32 lane * 16 elem
  if (idx >= 8 * 8 * 32 * 16) return;
  int e = idx & 15, lane = (idx >> 4) & 31, nt = (idx >> 9) & 7, kk = idx >> 12;
  int k = kk * 32 + (lane >> 4) * 8 + (e & 7) + (e >> 3) * 16;
  int n = nt * 16 + (lane & 15);
  bswz[idx] = (_Float16)W3[(size_t)k * 128 + n];
}

// ---- WMMA GEMM: hs3[NT,128] = tasks_f16[NT,256] @ W3_f16[256,128], f32 accumulate ----
// 8 waves/block; wave w owns N-tile w (16 cols); block owns 16 rows; 8 chained K=32 WMMAs.
__global__ __launch_bounds__(256) void k_wmma_gemm(const _Float16* __restrict__ tasks,
                                                   const _Float16* __restrict__ bswz,
                                                   float* __restrict__ hs3) {
  const int wave = threadIdx.x >> 5;
  const int lane = threadIdx.x & 31;
  const int m0 = blockIdx.x * 16;
  const int mr = lane & 15;
  const int kb = (lane >> 4) * 8;   // A 16-bit layout: lo lanes K {0..7,16..23}, hi lanes +8
  const _Float16* arow = tasks + (size_t)(m0 + mr) * 256;
  v8f c = {};
  #pragma unroll
  for (int kk = 0; kk < 8; ++kk) {
    h8 alo = *(const h8*)(arow + kk * 32 + kb);
    h8 ahi = *(const h8*)(arow + kk * 32 + kb + 16);
    const _Float16* bp = bswz + ((size_t)(kk * 8 + wave) * 32 + lane) * 16;
    h8 blo = *(const h8*)(bp);
    h8 bhi = *(const h8*)(bp + 8);
    v16h a, b;
    #pragma unroll
    for (int i = 0; i < 8; ++i) { a[i] = alo[i]; a[i + 8] = ahi[i]; b[i] = blo[i]; b[i + 8] = bhi[i]; }
    c = __builtin_amdgcn_wmma_f32_16x16x32_f16(false, a, false, b, (short)0, c, false, false);
  }
  const int n = wave * 16 + (lane & 15);
  const int mb = (lane >> 4) * 8;   // D layout: lanes 0-15 rows 0-7, lanes 16-31 rows 8-15
  #pragma unroll
  for (int r = 0; r < 8; ++r) hs3[(size_t)(m0 + mb + r) * 128 + n] = c[r];
}

// ---- final: head-mean of normalized conv3 output + b3, relu ----
__global__ void k_final(const float* __restrict__ acc, const float* __restrict__ den,
                        const float* __restrict__ b3, float* __restrict__ out) {
  int i = blockIdx.x * 256 + threadIdx.x;
  if (i >= NT_ * 32) return;
  int n = i >> 5, cc = i & 31;
  float s = 0.f;
  #pragma unroll
  for (int h = 0; h < 4; ++h)
    s += acc[(size_t)n * 128 + h * 32 + cc] / (den[(size_t)n * 4 + h] + 1e-16f);
  float v = 0.25f * s + b3[cc];
  out[i] = v > 0.f ? v : 0.f;
}

extern "C" void kernel_launch(void* const* d_in, const int* in_sizes, int n_in,
                              void* d_out, int out_size, void* d_ws, size_t ws_size,
                              hipStream_t stream) {
  const float* x_data = (const float*)d_in[0];
  const float* x_tasks = (const float*)d_in[1];
  const float* x_dev  = (const float*)d_in[2];
  const int*   ei_dt  = (const int*)d_in[3];
  const float* ea_dt  = (const float*)d_in[4];
  const int*   ei_vt  = (const int*)d_in[5];
  const float* ea_vt  = (const float*)d_in[6];
  const int*   ei_tt  = (const int*)d_in[7];
  const float* ea_tt  = (const float*)d_in[8];
  const float *Ws1 = (const float*)d_in[9],  *Wd1 = (const float*)d_in[10], *We1 = (const float*)d_in[11];
  const float *As1 = (const float*)d_in[12], *Ad1 = (const float*)d_in[13], *Ae1 = (const float*)d_in[14];
  const float *b1  = (const float*)d_in[15];
  const float *Ws2 = (const float*)d_in[16], *Wd2 = (const float*)d_in[17], *We2 = (const float*)d_in[18];
  const float *As2 = (const float*)d_in[19], *Ad2 = (const float*)d_in[20], *Ae2 = (const float*)d_in[21];
  const float *b2  = (const float*)d_in[22];
  const float *W3  = (const float*)d_in[23], *We3 = (const float*)d_in[24];
  const float *As3 = (const float*)d_in[25], *Ad3 = (const float*)d_in[26];
  const float *Ae3 = (const float*)d_in[27];
  const float *b3  = (const float*)d_in[28];
  float* out = (float*)d_out;

  // ---- workspace carve-out (phase reuse: out1acc->out3acc, out2acc->hs3) ----
  char* w = (char*)d_ws;
  size_t off = 0;
  auto alloc = [&](size_t bytes) -> void* {
    void* p = w + off;
    off = (off + bytes + 255) & ~(size_t)255;
    return p;
  };
  float* hs1      = (float*)alloc((size_t)NDAT_ * 128 * 4);
  float* hs2      = (float*)alloc((size_t)NDEV_ * 128 * 4);
  float* ssrc1    = (float*)alloc((size_t)NDAT_ * 4 * 4);
  float* sdst1    = (float*)alloc((size_t)NT_ * 4 * 4);
  float* ssrc2    = (float*)alloc((size_t)NDEV_ * 4 * 4);
  float* sdst2    = (float*)alloc((size_t)NT_ * 4 * 4);
  float* ssrc3    = (float*)alloc((size_t)NT_ * 4 * 4);
  float* sdst3    = (float*)alloc((size_t)NT_ * 4 * 4);
  float* wd_f1    = (float*)alloc(16 * 4 * 4);
  float* we_f1    = (float*)alloc(8 * 4 * 4);
  float* wd_f2    = (float*)alloc(16 * 4 * 4);
  float* we_f2    = (float*)alloc(6 * 4 * 4);
  float* we_f3    = (float*)alloc(4 * 4 * 4);
  _Float16* bswz  = (_Float16*)alloc(8 * 8 * 32 * 16 * 2);
  unsigned* amax  = (unsigned*)alloc((size_t)NT_ * 4 * 4);
  float* den      = (float*)alloc((size_t)NT_ * 4 * 4);
  float* out1acc  = (float*)alloc((size_t)NT_ * 128 * 4);   // later: out3acc
  float* out2acc  = (float*)alloc((size_t)NT_ * 128 * 4);   // later: hs3
  _Float16* t16   = (_Float16*)alloc((size_t)NT_ * 256 * 2);
  float* easum    = (float*)alloc((size_t)NT_ * 4 * 4);
  float* cnt      = (float*)alloc((size_t)NT_ * 4);
  float* eamean   = (float*)alloc((size_t)NT_ * 4 * 4);
  float* out3acc  = out1acc;
  float* hs3      = out2acc;
  (void)ws_size; (void)n_in; (void)in_sizes; (void)out_size;

  auto cdiv = [](long a, long b) { return (int)((a + b - 1) / b); };

  // ---- phase 0: zero accumulators (memset 0 is valid -inf for amax encoding) ----
  (void)hipMemsetAsync(amax, 0, (size_t)NT_ * 4 * 4, stream);
  (void)hipMemsetAsync(den, 0, (size_t)NT_ * 4 * 4, stream);
  (void)hipMemsetAsync(out1acc, 0, (size_t)NT_ * 128 * 4, stream);
  (void)hipMemsetAsync(out2acc, 0, (size_t)NT_ * 128 * 4, stream);
  (void)hipMemsetAsync(easum, 0, (size_t)NT_ * 4 * 4, stream);
  (void)hipMemsetAsync(cnt, 0, (size_t)NT_ * 4, stream);

  // ---- weight folds + W3 pack ----
  k_fold<<<1, 256, 0, stream>>>(Wd1, Ad1, wd_f1, 16);
  k_fold<<<1, 256, 0, stream>>>(We1, Ae1, we_f1, 8);
  k_fold<<<1, 256, 0, stream>>>(Wd2, Ad2, wd_f2, 16);
  k_fold<<<1, 256, 0, stream>>>(We2, Ae2, we_f2, 6);
  k_fold<<<1, 256, 0, stream>>>(We3, Ae3, we_f3, 4);
  k_pack_w3<<<cdiv(8 * 8 * 32 * 16, 256), 256, 0, stream>>>(W3, bswz);

  // ---- source projections + per-node attention dots ----
  k_lingemm<<<cdiv(NDAT_, 2), 256, 0, stream>>>(x_data, Ws1, hs1, NDAT_, 8);
  k_lingemm<<<cdiv(NDEV_, 2), 256, 0, stream>>>(x_dev, Ws2, hs2, NDEV_, 8);
  k_hdot<<<cdiv((long)NDAT_ * 4, 256), 256, 0, stream>>>(hs1, As1, ssrc1, NDAT_);
  k_hdot<<<1, 256, 0, stream>>>(hs2, As2, ssrc2, NDEV_);
  k_small<<<cdiv(NT_, 256), 256, 0, stream>>>(x_tasks, wd_f1, sdst1, NT_, 16);
  k_small<<<cdiv(NT_, 256), 256, 0, stream>>>(x_tasks, wd_f2, sdst2, NT_, 16);

  // ---- conv1 (data -> tasks) ----
  k_edge_max<<<cdiv(EDT_, 256), 256, 0, stream>>>(ei_dt, ei_dt + EDT_, ea_dt, 8, we_f1,
                                                  ssrc1, sdst1, amax, EDT_);
  k_edge_acc<<<cdiv((long)EDT_ * 32, 256), 256, 0, stream>>>(ei_dt, ei_dt + EDT_, ea_dt, 8, we_f1,
                                                             ssrc1, sdst1, amax, hs1,
                                                             out1acc, den, EDT_);
  k_norm<<<cdiv((long)NT_ * 128, 256), 256, 0, stream>>>(out1acc, den, b1, t16, 0);

  // ---- conv2 (devices -> tasks) ----
  (void)hipMemsetAsync(amax, 0, (size_t)NT_ * 4 * 4, stream);
  (void)hipMemsetAsync(den, 0, (size_t)NT_ * 4 * 4, stream);
  k_edge_max<<<cdiv(EVT_, 256), 256, 0, stream>>>(ei_vt, ei_vt + EVT_, ea_vt, 6, we_f2,
                                                  ssrc2, sdst2, amax, EVT_);
  k_edge_acc<<<cdiv((long)EVT_ * 32, 256), 256, 0, stream>>>(ei_vt, ei_vt + EVT_, ea_vt, 6, we_f2,
                                                             ssrc2, sdst2, amax, hs2,
                                                             out2acc, den, EVT_);
  k_norm<<<cdiv((long)NT_ * 128, 256), 256, 0, stream>>>(out2acc, den, b2, t16, 128);

  // ---- self-loop edge-attr mean ----
  k_easum<<<cdiv(ETT_, 256), 256, 0, stream>>>(ei_tt + ETT_, ea_tt, easum, cnt, ETT_);
  k_eamean<<<cdiv(NT_, 256), 256, 0, stream>>>(easum, cnt, eamean);

  // ---- conv3 projection via WMMA (tasks[NT,256] @ W3[256,128]) ----
  k_wmma_gemm<<<NT_ / 16, 256, 0, stream>>>(t16, bswz, hs3);
  k_hdot<<<cdiv((long)NT_ * 4, 256), 256, 0, stream>>>(hs3, As3, ssrc3, NT_);
  k_hdot<<<cdiv((long)NT_ * 4, 256), 256, 0, stream>>>(hs3, Ad3, sdst3, NT_);

  // ---- conv3 (tasks -> tasks with self loops) ----
  (void)hipMemsetAsync(amax, 0, (size_t)NT_ * 4 * 4, stream);
  (void)hipMemsetAsync(den, 0, (size_t)NT_ * 4 * 4, stream);
  (void)hipMemsetAsync(out3acc, 0, (size_t)NT_ * 128 * 4, stream);
  k_edge3_max<<<cdiv(ETT_ + NT_, 256), 256, 0, stream>>>(ei_tt, ei_tt + ETT_, ea_tt, eamean,
                                                         we_f3, ssrc3, sdst3, amax);
  k_edge3_acc<<<cdiv((long)(ETT_ + NT_) * 32, 256), 256, 0, stream>>>(ei_tt, ei_tt + ETT_, ea_tt,
                                                                      eamean, we_f3, ssrc3, sdst3,
                                                                      amax, hs3, out3acc, den);
  k_final<<<cdiv((long)NT_ * 32, 256), 256, 0, stream>>>(out3acc, den, b3, out);
}
